// BitLinear158_54683523612693
// MI455X (gfx1250) — compile-verified, matrix operand order checked by reading, not run on hardware
//
#include <hip/hip_runtime.h>
#include <hip/hip_bf16.h>
#include <stdint.h>

typedef __attribute__((ext_vector_type(16))) _Float16 v16h;
typedef __attribute__((ext_vector_type(8)))  _Float16 v8h;
typedef __attribute__((ext_vector_type(4)))  _Float16 v4h;
typedef __attribute__((ext_vector_type(8)))  float    v8f;

#define TILE_M 128
#define TILE_N 128

// ---------------------------------------------------------------------------
// CDNA5 async global->LDS copy (16B per lane), tracked by ASYNCcnt.
// vdst VGPR = LDS byte offset (generic shared ptr low 32 bits), vaddr = 64-bit
// global address. Inline asm per bridge doc (portable across toolchains).
// ---------------------------------------------------------------------------
__device__ __forceinline__ void async_copy_b128(uint32_t lds_byte_off, const void* gaddr) {
    asm volatile("global_load_async_to_lds_b128 %0, %1, off"
                 :: "v"(lds_byte_off), "v"(gaddr)
                 : "memory");
}
__device__ __forceinline__ void wait_asynccnt0() {
    asm volatile("s_wait_asynccnt 0x0" ::: "memory");
}
__device__ __forceinline__ uint32_t lds_off32(const void* p) {
    return (uint32_t)(uintptr_t)p;   // LDS aperture occupies addr[63:32]; low 32 = LDS offset
}

// ---------------------------------------------------------------------------
// Kernel 0: zero the gamma accumulator in workspace
// ---------------------------------------------------------------------------
__global__ void bl_init_kernel(float* ws) {
    if (threadIdx.x == 0) ws[0] = 0.0f;
}

// ---------------------------------------------------------------------------
// Kernel 1: sum(|W|) grid-stride reduction -> ws[0]
// ---------------------------------------------------------------------------
__global__ void bl_absum_kernel(const float* __restrict__ w, float* __restrict__ out, int n) {
    __shared__ float red[256];
    float s = 0.0f;
    for (int i = blockIdx.x * blockDim.x + threadIdx.x; i < n; i += gridDim.x * blockDim.x)
        s += __builtin_fabsf(w[i]);
    red[threadIdx.x] = s;
    __syncthreads();
    for (int off = 128; off > 0; off >>= 1) {
        if ((int)threadIdx.x < off) red[threadIdx.x] += red[threadIdx.x + off];
        __syncthreads();
    }
    if (threadIdx.x == 0) atomicAdd(out, red[0]);
}

__device__ __forceinline__ _Float16 tern_quant(float v, float thr) {
    return (_Float16)((__builtin_fabsf(v) >= thr) ? (v < 0.f ? -1.f : 1.f) : 0.f);
}

// ---------------------------------------------------------------------------
// Kernel 2a: pre-quantize W -> ternary f16 (one pass; 32 MB stays L2-resident)
// ---------------------------------------------------------------------------
__global__ void bl_quantw_kernel(const float* __restrict__ w, const float* __restrict__ gsum,
                                 _Float16* __restrict__ wh, int n4, float inv_nw) {
    const float gamma = gsum[0] * inv_nw + 1e-6f;
    const float thr   = 0.5f * gamma;
    for (int i = blockIdx.x * blockDim.x + threadIdx.x; i < n4; i += gridDim.x * blockDim.x) {
        float4 wv = ((const float4*)w)[i];
        v4h hw;
        hw[0] = tern_quant(wv.x, thr);
        hw[1] = tern_quant(wv.y, thr);
        hw[2] = tern_quant(wv.z, thr);
        hw[3] = tern_quant(wv.w, thr);
        ((v4h*)wh)[i] = hw;
    }
}

// ---------------------------------------------------------------------------
// Kernel 2b: pre-convert x -> f16
// ---------------------------------------------------------------------------
__global__ void bl_cvtx_kernel(const float* __restrict__ x, _Float16* __restrict__ xh, int n4) {
    for (int i = blockIdx.x * blockDim.x + threadIdx.x; i < n4; i += gridDim.x * blockDim.x) {
        float4 xv = ((const float4*)x)[i];
        v4h hx;
        hx[0] = (_Float16)xv.x; hx[1] = (_Float16)xv.y;
        hx[2] = (_Float16)xv.z; hx[3] = (_Float16)xv.w;
        ((v4h*)xh)[i] = hx;
    }
}

// ---------------------------------------------------------------------------
// Fragment loader: ISA 16-bit A/B matrix layout for 16x16x32 WMMA.
// Lanes 0-15: row = lane,    K = {0..7, 16..23}
// Lanes 16-31: row = lane-16, K = {8..15, 24..31}
// ---------------------------------------------------------------------------
__device__ __forceinline__ v16h load_frag(const _Float16* __restrict__ p) {
    v8h lo = *(const v8h*)(p);
    v8h hi = *(const v8h*)(p + 16);
    v16h r;
#pragma unroll
    for (int i = 0; i < 8; ++i) { r[i] = lo[i]; r[8 + i] = hi[i]; }
    return r;
}

// ---------------------------------------------------------------------------
// Kernel 3 (primary): f16 GEMM, double-buffered LDS filled by ASYNC global->LDS
// DMA. out[M,N] = xh[M,K] @ wh[N,K]^T. Block 256 thr (8 waves), 128x128xK64.
// Wave (wm=wid>>1, wn=wid&1) owns a 32x64 sub-tile -> 16 WMMAs / K-step.
// ---------------------------------------------------------------------------
#define TK 64
__global__ void __launch_bounds__(256)
bl_gemm_f16_kernel(const _Float16* __restrict__ xh, const _Float16* __restrict__ wh,
                   float* __restrict__ out, int M, int N, int K) {
    __shared__ _Float16 lA[2][TILE_M * TK];
    __shared__ _Float16 lB[2][TILE_N * TK];

    const int tid    = threadIdx.x;
    const int lane   = tid & 31;
    const int wid    = tid >> 5;
    const int wave_m = wid >> 1;
    const int wave_n = wid & 1;

    const int block_m = blockIdx.y * TILE_M;
    const int block_n = blockIdx.x * TILE_N;

    // staging map: chunk c = it*256+tid covers 8 halves (16B)
    int srow[4], scol[4];
#pragma unroll
    for (int it = 0; it < 4; ++it) {
        const int f = (it * 256 + tid) * 8;
        srow[it] = f >> 6;
        scol[it] = f & 63;
    }

    v8f acc[2][4];
#pragma unroll
    for (int i = 0; i < 2; ++i)
#pragma unroll
        for (int j = 0; j < 4; ++j)
#pragma unroll
            for (int g = 0; g < 8; ++g) acc[i][j][g] = 0.0f;

    const int k0   = (lane < 16) ? 0 : 8;
    const int frow = lane & 15;

    // async-stage tile (kk) into LDS buffer `buf`: 8 x b128 DMA per thread
    auto stage = [&](int kk, int buf) {
#pragma unroll
        for (int it = 0; it < 4; ++it) {
            async_copy_b128(lds_off32(&lA[buf][srow[it] * TK + scol[it]]),
                            xh + (size_t)(block_m + srow[it]) * K + kk + scol[it]);
            async_copy_b128(lds_off32(&lB[buf][srow[it] * TK + scol[it]]),
                            wh + (size_t)(block_n + srow[it]) * K + kk + scol[it]);
        }
    };

    const int nk = K / TK;
    stage(0, 0);
    wait_asynccnt0();
    __syncthreads();

    for (int t = 0; t < nk; ++t) {
        const int buf = t & 1;
        if (t + 1 < nk) stage((t + 1) * TK, (t + 1) & 1);  // DMA next tile

        // compute current tile: 2 K-fragments of 32 per TK=64
#pragma unroll
        for (int ks = 0; ks < 2; ++ks) {
            v16h afrag[2], bfrag[4];
#pragma unroll
            for (int im = 0; im < 2; ++im) {
                const int r = wave_m * 32 + im * 16 + frow;
                afrag[im] = load_frag(&lA[buf][r * TK + ks * 32 + k0]);
            }
#pragma unroll
            for (int jn = 0; jn < 4; ++jn) {
                const int r = wave_n * 64 + jn * 16 + frow;
                bfrag[jn] = load_frag(&lB[buf][r * TK + ks * 32 + k0]);
            }
#pragma unroll
            for (int im = 0; im < 2; ++im)
#pragma unroll
                for (int jn = 0; jn < 4; ++jn)
                    acc[im][jn] = __builtin_amdgcn_wmma_f32_16x16x32_f16(
                        false, afrag[im], false, bfrag[jn],
                        (short)0, acc[im][jn], false, false);
        }

        if (t + 1 < nk) wait_asynccnt0();   // DMA of next tile complete
        __syncthreads();
    }

    // epilogue: VGPR g holds (M = g + 8*(lane>=16), N = lane&15); NT stores so
    // the streamed 256MB output doesn't evict the L2-resident weights.
    const int crow_off = (lane >> 4) << 3;
    const int ccol     = lane & 15;
#pragma unroll
    for (int im = 0; im < 2; ++im)
#pragma unroll
        for (int jn = 0; jn < 4; ++jn) {
            const int base_r = block_m + wave_m * 32 + im * 16 + crow_off;
            const int base_c = block_n + wave_n * 64 + jn * 16 + ccol;
#pragma unroll
            for (int g = 0; g < 8; ++g)
                __builtin_nontemporal_store(acc[im][jn][g],
                                            out + (size_t)(base_r + g) * N + base_c);
        }
}

// ---------------------------------------------------------------------------
// Kernel 3-fallback: inline-conversion GEMM (TILE_K=32), used only when the
// workspace is too small for the f16 staging buffers.
// ---------------------------------------------------------------------------
#define TKF 32
__global__ void __launch_bounds__(256)
bl_gemm_inline_kernel(const float* __restrict__ x, const float* __restrict__ w,
                      const float* __restrict__ gsum, float* __restrict__ out,
                      int M, int N, int K, float inv_nw) {
    __shared__ _Float16 lA[TILE_M * TKF];
    __shared__ _Float16 lB[TILE_N * TKF];

    const int tid    = threadIdx.x;
    const int lane   = tid & 31;
    const int wid    = tid >> 5;
    const int wave_m = wid >> 1;
    const int wave_n = wid & 1;
    const int block_m = blockIdx.y * TILE_M;
    const int block_n = blockIdx.x * TILE_N;

    const float gamma = gsum[0] * inv_nw + 1e-6f;
    const float thr   = 0.5f * gamma;

    v8f acc[2][4];
#pragma unroll
    for (int i = 0; i < 2; ++i)
#pragma unroll
        for (int j = 0; j < 4; ++j)
#pragma unroll
            for (int g = 0; g < 8; ++g) acc[i][j][g] = 0.0f;

    const int k0   = (lane < 16) ? 0 : 8;
    const int frow = lane & 15;

    for (int kk = 0; kk < K; kk += TKF) {
        __syncthreads();
#pragma unroll
        for (int it = 0; it < 4; ++it) {
            const int f   = (it * 256 + tid) * 4;
            const int row = f >> 5;
            const int col = f & 31;

            float4 xv = *(const float4*)(x + (size_t)(block_m + row) * K + kk + col);
            v4h hx;
            hx[0] = (_Float16)xv.x; hx[1] = (_Float16)xv.y;
            hx[2] = (_Float16)xv.z; hx[3] = (_Float16)xv.w;
            *(v4h*)(&lA[row * TKF + col]) = hx;

            float4 wv = *(const float4*)(w + (size_t)(block_n + row) * K + kk + col);
            v4h hw;
            hw[0] = tern_quant(wv.x, thr);
            hw[1] = tern_quant(wv.y, thr);
            hw[2] = tern_quant(wv.z, thr);
            hw[3] = tern_quant(wv.w, thr);
            *(v4h*)(&lB[row * TKF + col]) = hw;
        }
        __syncthreads();

        v16h afrag[2], bfrag[4];
#pragma unroll
        for (int im = 0; im < 2; ++im)
            afrag[im] = load_frag(&lA[(wave_m * 32 + im * 16 + frow) * TKF + k0]);
#pragma unroll
        for (int jn = 0; jn < 4; ++jn)
            bfrag[jn] = load_frag(&lB[(wave_n * 64 + jn * 16 + frow) * TKF + k0]);
#pragma unroll
        for (int im = 0; im < 2; ++im)
#pragma unroll
            for (int jn = 0; jn < 4; ++jn)
                acc[im][jn] = __builtin_amdgcn_wmma_f32_16x16x32_f16(
                    false, afrag[im], false, bfrag[jn],
                    (short)0, acc[im][jn], false, false);
    }

    const int crow_off = (lane >> 4) << 3;
    const int ccol     = lane & 15;
#pragma unroll
    for (int im = 0; im < 2; ++im)
#pragma unroll
        for (int jn = 0; jn < 4; ++jn) {
            const int base_r = block_m + wave_m * 32 + im * 16 + crow_off;
            const int base_c = block_n + wave_n * 64 + jn * 16 + ccol;
#pragma unroll
            for (int g = 0; g < 8; ++g)
                __builtin_nontemporal_store(acc[im][jn][g],
                                            out + (size_t)(base_r + g) * N + base_c);
        }
}

// ---------------------------------------------------------------------------
extern "C" void kernel_launch(void* const* d_in, const int* in_sizes, int n_in,
                              void* d_out, int out_size, void* d_ws, size_t ws_size,
                              hipStream_t stream) {
    const float* x = (const float*)d_in[0];
    const float* w = (const float*)d_in[1];
    float* out     = (float*)d_out;

    const int K  = 4096;
    const int N  = 4096;
    const int M  = in_sizes[0] / K;      // 16384
    const int NW = in_sizes[1];          // 4096*4096
    const float inv_nw = 1.0f / (float)NW;

    float* gsum = (float*)d_ws;
    bl_init_kernel<<<1, 32, 0, stream>>>(gsum);
    bl_absum_kernel<<<1024, 256, 0, stream>>>(w, gsum, NW);

    const size_t wh_off  = 256;
    const size_t wh_size = (size_t)N * K * sizeof(_Float16);      // 32 MB
    const size_t xh_off  = wh_off + wh_size;
    const size_t xh_size = (size_t)M * K * sizeof(_Float16);      // 128 MB
    const size_t need    = xh_off + xh_size;

    dim3 grid(N / TILE_N, M / TILE_M);   // (32, 128)

    if (ws_size >= need) {
        _Float16* wh = (_Float16*)((char*)d_ws + wh_off);
        _Float16* xh = (_Float16*)((char*)d_ws + xh_off);
        bl_quantw_kernel<<<2048, 256, 0, stream>>>(w, gsum, wh, NW / 4, inv_nw);
        bl_cvtx_kernel<<<2048, 256, 0, stream>>>(x, xh, in_sizes[0] / 4);
        bl_gemm_f16_kernel<<<grid, 256, 0, stream>>>(xh, wh, out, M, N, K);
    } else {
        bl_gemm_inline_kernel<<<grid, 256, 0, stream>>>(x, w, gsum, out, M, N, K, inv_nw);
    }
}
